// GCNEncoder_61014305407122
// MI455X (gfx1250) — matrix operand order repaired and checked by reading, not compiled, not run on hardware
//
#include <hip/hip_runtime.h>
#include <math.h>

typedef __attribute__((ext_vector_type(16))) _Float16 v16h;
typedef __attribute__((ext_vector_type(8)))  _Float16 v8h;
typedef __attribute__((ext_vector_type(8)))  float    v8f;

#define Bb  32
#define Cc  64
#define Fd  128
#define TPn 32
#define Gn  (Bb*TPn)    // 1024 graphs
#define N1  (Gn*Cc)     // 65536 nodes layer 1 (also padded layer-2 node count)
#define En  512
#define NH  256
#define NO  128
#define K1n 52          // ceil(0.8*64)
#define K2n 42          // ceil(0.8*52)
#define M2n (Gn*K1n)    // 53248 real layer-2 nodes

// ---- fragment helper: 16 halves from two aligned 8-half chunks -----------
__device__ __forceinline__ v16h load_frag(const _Float16* p0, const _Float16* p1) {
  v8h lo = *(const v8h*)p0, hi = *(const v8h*)p1;
  return __builtin_shufflevector(lo, hi, 0,1,2,3,4,5,6,7,8,9,10,11,12,13,14,15);
}

// ---- permute h (B,C,F,TP) -> x0h (G*C, F) f16 row-major ------------------
__global__ void k_permute(const float* __restrict__ h, _Float16* __restrict__ x0) {
  int idx = blockIdx.x * blockDim.x + threadIdx.x;
  if (idx >= Bb*Cc*Fd*TPn) return;
  int tp = idx & (TPn - 1);
  int f  = (idx >> 5) & (Fd - 1);
  int c  = (idx >> 12) & (Cc - 1);
  int b  = idx >> 18;
  x0[(((size_t)(b*TPn + tp)*Cc + c)*Fd) + f] = (_Float16)h[idx];
}

// ---- weight transpose+convert: W[K][N] f32 -> BT[N][K] f16 ---------------
__global__ void k_prepBT(const float* __restrict__ W, _Float16* __restrict__ BT,
                         int K, int N) {
  int idx = blockIdx.x * blockDim.x + threadIdx.x;
  if (idx >= K*N) return;
  int k = idx / N, n = idx - k*N;
  BT[(size_t)n*K + k] = (_Float16)W[idx];
}

// ---- template adjacency (shared across all graphs) -----------------------
__global__ void k_tmpl_init(float* A1w, float* A1b) {
  int i = blockIdx.x * blockDim.x + threadIdx.x;
  if (i >= 64*64) return;
  float v = ((i >> 6) == (i & 63)) ? 1.0f : 0.0f;   // self-loop on diagonal
  A1w[i] = v; A1b[i] = v;
}

__global__ void k_tmpl_scatter(const int* __restrict__ ei, const float* __restrict__ ew,
                               float* A1w, float* A1b) {
  int e = blockIdx.x * blockDim.x + threadIdx.x;
  if (e >= En) return;
  int r = ei[e], c = ei[En + e];
  atomicAdd(&A1w[c*64 + r], ew[e]);
  atomicAdd(&A1b[c*64 + r], 1.0f);
}

// normalize; A1w -> f16 copy (WMMA A-operand), A1b normalized in place (f32)
__global__ void k_tmpl_norm(const float* __restrict__ A1w, float* A1b,
                            _Float16* __restrict__ A1wh) {
  __shared__ float dw[64], db[64];
  int c = threadIdx.x;
  float sw = 0.f, sb = 0.f;
  for (int r = 0; r < 64; ++r) { sw += A1w[c*64 + r]; sb += A1b[c*64 + r]; }
  dw[c] = rsqrtf(sw); db[c] = rsqrtf(sb);
  __syncthreads();
  for (int r = 0; r < 64; ++r) {
    A1wh[c*64 + r] = (_Float16)(A1w[c*64 + r] * dw[r] * dw[c]);
    A1b[c*64 + r] *= db[r] * db[c];
  }
}

// ---- WMMA GEMM: A[M][K] f16 row-major  @  Bt[N][K] f16 (K-major) ---------
// output Ct in per-graph B-layout: f16 [g=row/64][n][k=row%64] so the next
// aggregation kernel consumes it as its B operand with b128 fragment loads.
// one wave per 16(M)x64(N) block, 4 WMMAs per A fragment.
__global__ void k_gemm_wmma(const _Float16* __restrict__ A, const _Float16* __restrict__ Bt,
                            _Float16* __restrict__ Ct, int M, int N, int K) {
  int wave = blockIdx.x * (blockDim.x >> 5) + (threadIdx.x >> 5);
  int lane = threadIdx.x & 31;
  int nb = N >> 6;
  int tm = wave / nb, tnb = wave - tm * nb;
  if (tm * 16 >= M) return;                 // wave-uniform guard
  int ml = lane & 15, hh = lane >> 4;
  const _Float16* arow = A + (size_t)(tm*16 + ml) * K;
  int n0 = tnb*64 + ml;
  v8f acc[4] = {};
  for (int k0 = 0; k0 < K; k0 += 32) {
    v16h a = load_frag(arow + k0 + 8*hh, arow + k0 + 16 + 8*hh);
#pragma unroll
    for (int s = 0; s < 4; ++s) {
      const _Float16* bp = Bt + (size_t)(n0 + s*16)*K + k0 + 8*hh;
      v16h b = load_frag(bp, bp + 16);
      acc[s] = __builtin_amdgcn_wmma_f32_16x16x32_f16(false, a, false, b,
                                                      (short)0, acc[s], false, false);
    }
  }
  int g = (tm*16) >> 6;
  int kloc = ((tm*16) & 63) + 8*hh;          // rows r=0..7 are contiguous in k
#pragma unroll
  for (int s = 0; s < 4; ++s) {
    v8h o;
#pragma unroll
    for (int r = 0; r < 8; ++r) o[r] = (_Float16)acc[s][r];
    *(v8h*)(Ct + ((size_t)g*N + n0 + s*16)*64 + kloc) = o;   // b128 store
  }
}

// ---- batched WMMA aggregation + bias + BN + PReLU epilogue ---------------
// out_g(64 x N) = Adj(64x64 f16) @ XW_g(64 x N); XWt in [g][n][k] layout.
template <int NCOLS, bool PER_GRAPH_ADJ, typename OutT>
__global__ void k_amult_wmma(const _Float16* __restrict__ Adj, const _Float16* __restrict__ XWt,
                             const float* __restrict__ gb, const float* __restrict__ gamma,
                             const float* __restrict__ beta, const float* __restrict__ mean,
                             const float* __restrict__ var, const float* __restrict__ pa,
                             OutT* __restrict__ xout) {
  const int WPG = (NCOLS/64) * 4;           // waves per graph: 16 (256) or 8 (128)
  int wave = blockIdx.x * (blockDim.x >> 5) + (threadIdx.x >> 5);
  int lane = threadIdx.x & 31;
  int g = wave / WPG, t = wave - g*WPG;
  int tm = t / (NCOLS/64), tnb = t - tm*(NCOLS/64);
  int ml = lane & 15, hh = lane >> 4;
  const _Float16* arow = Adj + (PER_GRAPH_ADJ ? (size_t)g*64*64 : (size_t)0)
                             + (size_t)(tm*16 + ml)*64;
  int n0 = tnb*64 + ml;
  float aslope = pa[0];
  v8f acc[4] = {};
#pragma unroll
  for (int k0 = 0; k0 < 64; k0 += 32) {
    v16h a = load_frag(arow + k0 + 8*hh, arow + k0 + 16 + 8*hh);
#pragma unroll
    for (int s = 0; s < 4; ++s) {
      const _Float16* bp = XWt + ((size_t)g*NCOLS + n0 + s*16)*64 + k0 + 8*hh;
      v16h b = load_frag(bp, bp + 16);
      acc[s] = __builtin_amdgcn_wmma_f32_16x16x32_f16(false, a, false, b,
                                                      (short)0, acc[s], false, false);
    }
  }
#pragma unroll
  for (int s = 0; s < 4; ++s) {
    int n = n0 + s*16;
    float bias = gb[n], ga = gamma[n], be = beta[n], mu = mean[n];
    float inv = rsqrtf(var[n] + 1e-5f);
#pragma unroll
    for (int r = 0; r < 8; ++r) {
      int row = tm*16 + r + 8*hh;
      float y = acc[s][r] + bias;
      y = ga * (y - mu) * inv + be;
      y = (y >= 0.f) ? y : aslope * y;
      xout[((size_t)g*64 + row)*NCOLS + n] = (OutT)y;   // coalesced across lanes
    }
  }
}

// ---- row dots ------------------------------------------------------------
__global__ void k_dot_f16(const _Float16* __restrict__ x, const float* __restrict__ w,
                          float* __restrict__ t, int n, int d) {
  int i = blockIdx.x * blockDim.x + threadIdx.x;
  if (i >= n) return;
  const v8h* row = (const v8h*)(x + (size_t)i * d);
  float acc = 0.f;
  for (int c = 0; c < d/8; ++c) {
    v8h v = row[c];
#pragma unroll
    for (int u = 0; u < 8; ++u) acc += (float)v[u] * w[c*8 + u];
  }
  t[i] = acc;
}

__global__ void k_dot_f32(const float* __restrict__ x, const float* __restrict__ w,
                          float* __restrict__ t, int n, int d) {
  int i = blockIdx.x * blockDim.x + threadIdx.x;
  if (i >= n) return;
  const float* row = x + (size_t)i * d;
  float acc = 0.f;
  for (int l = 0; l < d; ++l) acc += row[l] * w[l];
  t[i] = acc;
}

// ---- pool1 score: sc = A1b @ t + b per graph -----------------------------
__global__ void k_score1(const float* __restrict__ A1b, const float* __restrict__ t1,
                         const float* __restrict__ pb, float* __restrict__ sc) {
  __shared__ float ts[64];
  int g = blockIdx.x, c = threadIdx.x;
  ts[c] = t1[g*64 + c];
  __syncthreads();
  float acc = pb[0];
  for (int r = 0; r < 64; ++r) acc += A1b[c*64 + r] * ts[r];
  sc[g*64 + c] = acc;
}

// ---- stable top-k1 (desc, ties -> lower idx), gate by tanh ---------------
// xk1h written PADDED f16 row-major (row g*64+rank); nm = local rank or -1.
__global__ void k_topk1(const float* __restrict__ sc, const _Float16* __restrict__ x1,
                        _Float16* __restrict__ xk1, int* __restrict__ nm) {
  __shared__ float s[64];
  int g = blockIdx.x, c = threadIdx.x;
  float v = sc[g*64 + c];
  s[c] = v; __syncthreads();
  int rank = 0;
  for (int r = 0; r < 64; ++r) { float o = s[r]; rank += (o > v) || (o == v && r < c); }
  int m = -1;
  if (rank < K1n) {
    m = rank;
    float gate = tanhf(v);
    const v8h* src = (const v8h*)(x1 + ((size_t)g*64 + c)*NH);
    v8h* dst = (v8h*)(xk1 + ((size_t)g*64 + rank)*NH);
    for (int ch = 0; ch < NH/8; ++ch) {
      v8h in = src[ch], o;
#pragma unroll
      for (int u = 0; u < 8; ++u) o[u] = (_Float16)(gate * (float)in[u]);
      dst[ch] = o;
    }
  }
  nm[g*64 + c] = m;
}

// zero the 12 pad rows per graph of padded xk1h (v8h stores)
__global__ void k_padzero(_Float16* __restrict__ xk1) {
  int idx = blockIdx.x * blockDim.x + threadIdx.x;   // Gn*12*NH/8 chunks
  if (idx >= Gn * 12 * (NH/8)) return;
  int ch = idx & (NH/8 - 1);
  int rem = idx / (NH/8);
  int p = rem % 12, g = rem / 12;
  v8h z = {};
  *(v8h*)(xk1 + ((size_t)g*64 + K1n + p)*NH + ch*8) = z;
}

// ---- per-graph layer2 adjacency (PADDED 64x64, real block 52x52) ---------
__global__ void k_adj2_init(float* A2w, float* A2b, _Float16* A2wh) {
  int g = blockIdx.x;
  size_t base = (size_t)g * 64 * 64;
  for (int i = threadIdx.x; i < 64*64; i += blockDim.x) {
    int c = i >> 6, r = i & 63;
    float v = (c == r && c < K1n) ? 1.0f : 0.0f;
    A2w[base + i] = v; A2b[base + i] = v;
    A2wh[base + i] = (_Float16)0.f;          // real block overwritten by norm
  }
}

__global__ void k_adj2_scatter(const int* __restrict__ ei, const float* __restrict__ ew,
                               const int* __restrict__ nm, float* A2w, float* A2b) {
  int tid = blockIdx.x * blockDim.x + threadIdx.x;
  if (tid >= Gn * En) return;
  int g = tid >> 9, e = tid & 511;
  int rl = nm[g*64 + ei[e]];
  int cl = nm[g*64 + ei[En + e]];
  if (rl >= 0 && cl >= 0) {
    size_t base = (size_t)g * 64 * 64;
    atomicAdd(&A2w[base + cl*64 + rl], ew[e]);
    atomicAdd(&A2b[base + cl*64 + rl], 1.0f);
  }
}

// normalize; A2w -> f16 (WMMA A-operand), A2b normalized in place (f32)
__global__ void k_adj2_norm(const float* __restrict__ A2w, float* A2b,
                            _Float16* __restrict__ A2wh) {
  __shared__ float dw[K1n], db[K1n];
  int g = blockIdx.x, c = threadIdx.x;
  size_t base = (size_t)g * 64 * 64;
  if (c < K1n) {
    float sw = 0.f, sb = 0.f;
    for (int r = 0; r < K1n; ++r) { sw += A2w[base + c*64 + r]; sb += A2b[base + c*64 + r]; }
    dw[c] = rsqrtf(sw); db[c] = rsqrtf(sb);
  }
  __syncthreads();
  if (c < K1n) {
    for (int r = 0; r < K1n; ++r) {
      A2wh[base + c*64 + r] = (_Float16)(A2w[base + c*64 + r] * dw[r] * dw[c]);
      A2b[base + c*64 + r] *= db[r] * db[c];
    }
  }
}

// ---- pool2 score (padded adjacency stride 64) ----------------------------
__global__ void k_score2(const float* __restrict__ A2b, const float* __restrict__ t2,
                         const float* __restrict__ pb, float* __restrict__ sc) {
  __shared__ float ts[K1n];
  int g = blockIdx.x, c = threadIdx.x;
  if (c < K1n) ts[c] = t2[g*64 + c];
  __syncthreads();
  if (c < K1n) {
    size_t base = (size_t)g * 64 * 64;
    float acc = pb[0];
    for (int r = 0; r < K1n; ++r) acc += A2b[base + c*64 + r] * ts[r];
    sc[g*K1n + c] = acc;
  }
}

__global__ void k_topk2(const float* __restrict__ sc, const float* __restrict__ x2,
                        float* __restrict__ outX) {
  __shared__ float s[K1n];
  int g = blockIdx.x, c = threadIdx.x;
  float v = 0.f;
  if (c < K1n) { v = sc[g*K1n + c]; s[c] = v; }
  __syncthreads();
  if (c < K1n) {
    int rank = 0;
    for (int r = 0; r < K1n; ++r) { float o = s[r]; rank += (o > v) || (o == v && r < c); }
    if (rank < K2n) {
      float gate = tanhf(v);
      const float* src = x2 + ((size_t)g*64 + c)*NO;   // padded layout
      float* dst = outX + ((size_t)g*K2n + rank)*NO;
      for (int jj = 0; jj < NO; ++jj) dst[jj] = src[jj] * gate;
    }
  }
}

__global__ void k_zmean(const float* __restrict__ outX, float* __restrict__ z) {
  int g = blockIdx.x, j = threadIdx.x;
  float acc = 0.f;
  for (int r = 0; r < K2n; ++r) acc += outX[((size_t)g*K2n + r)*NO + j];
  z[(size_t)g*NO + j] = acc * (1.0f / (float)K2n);
}

extern "C" void kernel_launch(void* const* d_in, const int* in_sizes, int n_in,
                              void* d_out, int out_size, void* d_ws, size_t ws_size,
                              hipStream_t stream) {
  (void)in_sizes; (void)n_in; (void)out_size; (void)ws_size;
  const float* h     = (const float*)d_in[0];
  const int*   ei    = (const int*)d_in[1];
  const float* ew    = (const float*)d_in[2];
  const float* gc1_w = (const float*)d_in[3];
  const float* gc1_b = (const float*)d_in[4];
  const float* bn1_g = (const float*)d_in[5];
  const float* bn1_b = (const float*)d_in[6];
  const float* bn1_m = (const float*)d_in[7];
  const float* bn1_v = (const float*)d_in[8];
  const float* pa1   = (const float*)d_in[9];
  const float* p1w   = (const float*)d_in[10];
  const float* p1b   = (const float*)d_in[11];
  const float* gc2_w = (const float*)d_in[12];
  const float* gc2_b = (const float*)d_in[13];
  const float* bn2_g = (const float*)d_in[14];
  const float* bn2_b = (const float*)d_in[15];
  const float* bn2_m = (const float*)d_in[16];
  const float* bn2_v = (const float*)d_in[17];
  const float* pa2   = (const float*)d_in[18];
  const float* p2w   = (const float*)d_in[19];
  const float* p2b   = (const float*)d_in[20];

  char* base = (char*)d_ws;
  auto alloc = [&](size_t bytes) { char* p = base; base += (bytes + 255) & ~(size_t)255; return p; };
  _Float16* x0h  = (_Float16*)alloc((size_t)N1 * Fd * 2);
  _Float16* BT1  = (_Float16*)alloc((size_t)Fd * NH * 2);
  _Float16* BT2  = (_Float16*)alloc((size_t)NH * NO * 2);
  _Float16* XW1t = (_Float16*)alloc((size_t)N1 * NH * 2);
  _Float16* x1h  = (_Float16*)alloc((size_t)N1 * NH * 2);
  _Float16* xk1h = (_Float16*)alloc((size_t)N1 * NH * 2);   // padded 64 rows/graph
  _Float16* XW2t = (_Float16*)alloc((size_t)N1 * NO * 2);
  _Float16* A1wh = (_Float16*)alloc(64*64*2);
  _Float16* A2wh = (_Float16*)alloc((size_t)Gn * 64*64 * 2);
  float* A1w = (float*)alloc(64*64*4);
  float* A1b = (float*)alloc(64*64*4);
  float* A2w = (float*)alloc((size_t)Gn * 64*64 * 4);
  float* A2b = (float*)alloc((size_t)Gn * 64*64 * 4);
  float* t1  = (float*)alloc((size_t)N1 * 4);
  float* sc1 = (float*)alloc((size_t)N1 * 4);
  int*   nm1 = (int*)alloc((size_t)N1 * 4);
  float* t2  = (float*)alloc((size_t)N1 * 4);
  float* sc2 = (float*)alloc((size_t)M2n * 4);
  float* x2p = (float*)alloc((size_t)N1 * NO * 4);          // padded f32

  float* outX = (float*)d_out;
  float* outZ = outX + (size_t)Gn * K2n * NO;

  k_permute<<<(N1*Fd + 255)/256, 256, 0, stream>>>(h, x0h);
  k_prepBT<<<(Fd*NH + 255)/256, 256, 0, stream>>>(gc1_w, BT1, Fd, NH);
  k_prepBT<<<(NH*NO + 255)/256, 256, 0, stream>>>(gc2_w, BT2, NH, NO);
  k_tmpl_init<<<16, 256, 0, stream>>>(A1w, A1b);
  k_tmpl_scatter<<<2, 256, 0, stream>>>(ei, ew, A1w, A1b);
  k_tmpl_norm<<<1, 64, 0, stream>>>(A1w, A1b, A1wh);

  // GEMM1: XW1t = x0h @ gc1_w  (M=65536,K=128,N=256): 4096*4 waves -> 2048 blocks
  k_gemm_wmma<<<(N1/16)*(NH/64)/8, 256, 0, stream>>>(x0h, BT1, XW1t, N1, NH, Fd);
  // aggregation l1 (WMMA): 1024*16 waves -> 2048 blocks; x1h f16 row-major
  k_amult_wmma<NH, false, _Float16><<<Gn*16/8, 256, 0, stream>>>(
      A1wh, XW1t, gc1_b, bn1_g, bn1_b, bn1_m, bn1_v, pa1, x1h);

  k_dot_f16<<<(N1 + 255)/256, 256, 0, stream>>>(x1h, p1w, t1, N1, NH);
  k_score1<<<Gn, 64, 0, stream>>>(A1b, t1, p1b, sc1);
  k_topk1<<<Gn, 64, 0, stream>>>(sc1, x1h, xk1h, nm1);
  k_padzero<<<(Gn*12*(NH/8) + 255)/256, 256, 0, stream>>>(xk1h);

  k_adj2_init<<<Gn, 256, 0, stream>>>(A2w, A2b, A2wh);
  k_adj2_scatter<<<(Gn*En + 255)/256, 256, 0, stream>>>(ei, ew, nm1, A2w, A2b);
  k_adj2_norm<<<Gn, 64, 0, stream>>>(A2w, A2b, A2wh);

  // GEMM2 (padded M): XW2t = xk1h @ gc2_w (M=65536,K=256,N=128): 4096*2 -> 1024 blocks
  k_gemm_wmma<<<(N1/16)*(NO/64)/8, 256, 0, stream>>>(xk1h, BT2, XW2t, N1, NO, NH);
  // aggregation l2 (WMMA, per-graph padded adjacency): 1024*8 waves -> 1024 blocks
  k_amult_wmma<NO, true, float><<<Gn*8/8, 256, 0, stream>>>(
      A2wh, XW2t, gc2_b, bn2_g, bn2_b, bn2_m, bn2_v, pa2, x2p);

  k_dot_f32<<<(N1 + 255)/256, 256, 0, stream>>>(x2p, p2w, t2, N1, NO);
  k_score2<<<Gn, 64, 0, stream>>>(A2b, t2, p2b, sc2);
  k_topk2<<<Gn, 64, 0, stream>>>(sc2, x2p, outX);
  k_zmean<<<Gn, NO, 0, stream>>>(outX, outZ);
}
